// EvoformerPairBlock_32779190403857
// MI455X (gfx1250) — matrix operand order repaired and checked by reading, not compiled, not run on hardware
//
#include <hip/hip_runtime.h>
#include <hip/hip_bf16.h>
#include <math.h>

// ---------------------------------------------------------------------------
// Types & helpers
// ---------------------------------------------------------------------------
typedef __bf16 bf16x16 __attribute__((ext_vector_type(16)));
typedef __bf16 bf16x8  __attribute__((ext_vector_type(8)));
typedef float  floatx8 __attribute__((ext_vector_type(8)));

#define DEV static __device__ __forceinline__

constexpr int NRES  = 256;
constexpr int MROWS = NRES * NRES;   // 65536 pair rows
constexpr int CS    = 384;
constexpr int CH    = 128;

DEV float bf2f(__bf16 b) {
  union { unsigned short s; __bf16 b; } i; i.b = b;
  union { unsigned u; float f; } o; o.u = ((unsigned)i.s) << 16;
  return o.f;
}
DEV __bf16 f2bf(float f) {
  union { float f; unsigned u; } i; i.f = f;
  unsigned r = (i.u + 0x7FFFu + ((i.u >> 16) & 1u)) >> 16;
  union { unsigned short s; __bf16 b; } o; o.s = (unsigned short)r;
  return o.b;
}
DEV float sigmoidf_(float x) { return 1.f / (1.f + __expf(-x)); }

// A-matrix fragment (16-bit, 16x32): per ISA, half0 lanes hold K {0..7,16..23},
// half1 lanes hold K {8..15,24..31}. Caller pre-offsets p by half*8 elements.
DEV bf16x16 load_fragA(const __bf16* p) {
  bf16x8 lo = *(const bf16x8*)(p);
  bf16x8 hi = *(const bf16x8*)(p + 16);
  return __builtin_shufflevector(lo, hi, 0,1,2,3,4,5,6,7,8,9,10,11,12,13,14,15);
}
// B-matrix fragment (16-bit, 32x16): half0 holds K {0..7,16..23}, half1 K {8..15,24..31}
// (same as A). Caller pre-offsets p by half*8 elements.
DEV bf16x16 load_fragB(const __bf16* p) {
  bf16x8 lo = *(const bf16x8*)(p);
  bf16x8 hi = *(const bf16x8*)(p + 16);
  return __builtin_shufflevector(lo, hi, 0,1,2,3,4,5,6,7,8,9,10,11,12,13,14,15);
}
DEV floatx8 wmma_bf16(bf16x16 a, bf16x16 b, floatx8 c) {
  return __builtin_amdgcn_wmma_f32_16x16x32_bf16(false, a, false, b, (short)0, c,
                                                 false, false);
}

// Register-blocked 32x32 wave tile (2x2 of 16x16): 4 WMMAs reuse 2 A-frags and
// 2 B-frags per K-step -> 2 b128 loads per WMMA instead of 4.
struct Acc4 { floatx8 a00, a01, a10, a11; };

DEV Acc4 mma_loop2(const __bf16* A, int ldA, int rowBase,
                   const __bf16* Bt, int ldB, int colBase, int K) {
  int lane = threadIdx.x & 31;
  int half = lane >> 4, l16 = lane & 15;
  const __bf16* pa0 = A  + (size_t)(rowBase + l16) * ldA + half * 8;
  const __bf16* pa1 = pa0 + (size_t)16 * ldA;
  const __bf16* pb0 = Bt + (size_t)(colBase + l16) * ldB + half * 8;
  const __bf16* pb1 = pb0 + (size_t)16 * ldB;
  floatx8 z = {0.f,0.f,0.f,0.f,0.f,0.f,0.f,0.f};
  Acc4 acc; acc.a00 = z; acc.a01 = z; acc.a10 = z; acc.a11 = z;
  for (int k0 = 0; k0 < K; k0 += 32) {
    bf16x16 a0 = load_fragA(pa0 + k0);
    bf16x16 a1 = load_fragA(pa1 + k0);
    bf16x16 b0 = load_fragB(pb0 + k0);
    bf16x16 b1 = load_fragB(pb1 + k0);
    acc.a00 = wmma_bf16(a0, b0, acc.a00);
    acc.a01 = wmma_bf16(a0, b1, acc.a01);
    acc.a10 = wmma_bf16(a1, b0, acc.a10);
    acc.a11 = wmma_bf16(a1, b1, acc.a11);
  }
  return acc;
}
DEV floatx8 acc_sel(const Acc4& a, int t) {
  return (t == 0) ? a.a00 : (t == 1) ? a.a01 : (t == 2) ? a.a10 : a.a11;
}

// ---------------------------------------------------------------------------
// Utility kernels
// ---------------------------------------------------------------------------
__global__ __launch_bounds__(256)
void wt_cast_k(const float* __restrict__ w, __bf16* __restrict__ o, int In, int Out) {
  int idx = blockIdx.x * 256 + threadIdx.x;
  if (idx >= In * Out) return;
  int oo = idx / In, ii = idx - oo * In;
  o[(size_t)oo * In + ii] = f2bf(w[(size_t)ii * Out + oo]);   // store transposed
}

__global__ __launch_bounds__(128)
void ln_bf16_k(const float* __restrict__ x, const float* __restrict__ g,
               const float* __restrict__ b, __bf16* __restrict__ out, int C) {
  int row = blockIdx.x;
  const float* xr = x + (size_t)row * C;
  float s = 0.f, s2 = 0.f;
  for (int c = threadIdx.x; c < C; c += 128) { float v = xr[c]; s += v; s2 += v * v; }
  __shared__ float rs[128], rq[128];
  rs[threadIdx.x] = s; rq[threadIdx.x] = s2; __syncthreads();
  for (int o = 64; o > 0; o >>= 1) {
    if (threadIdx.x < o) { rs[threadIdx.x] += rs[threadIdx.x + o];
                           rq[threadIdx.x] += rq[threadIdx.x + o]; }
    __syncthreads();
  }
  float mean = rs[0] / (float)C;
  float var  = rq[0] / (float)C - mean * mean;
  float inv  = rsqrtf(var + 1e-5f);
  for (int c = threadIdx.x; c < C; c += 128) {
    float v = (xr[c] - mean) * inv * g[c] + b[c];
    out[(size_t)row * C + c] = f2bf(v);
  }
}

__global__ __launch_bounds__(256)
void transpose_z_k(const float* __restrict__ s, float* __restrict__ d) {
  size_t idx = (size_t)blockIdx.x * 256 + threadIdx.x;   // over MROWS*CS
  int c = (int)(idx % CS);
  size_t r = idx / CS;
  int i = (int)(r >> 8), j = (int)(r & 255);
  d[((size_t)((j << 8) | i)) * CS + c] = s[idx];
}

__global__ __launch_bounds__(256)
void mask_t_k(const float* __restrict__ s, float* __restrict__ d) {
  int idx = blockIdx.x * 256 + threadIdx.x;   // 65536
  int i = idx >> 8, k = idx & 255;
  d[idx] = s[(k << 8) | i];
}

// tri_b = zl @ b_w  ([M,384] x [384,4]) -- tiny; scalar path
__global__ __launch_bounds__(256)
void proj4_k(const __bf16* __restrict__ lnz, const float* __restrict__ w,
             float* __restrict__ out) {
  int idx = blockIdx.x * 256 + threadIdx.x;   // MROWS*4
  int r = idx >> 2, h = idx & 3;
  const __bf16* xr = lnz + (size_t)r * CS;
  float s = 0.f;
  for (int c = 0; c < CS; ++c) s += bf2f(xr[c]) * w[c * 4 + h];
  out[idx] = s;
}

// ---------------------------------------------------------------------------
// GEMM kernels: block = 256 thr = 8 waves (4x2), block tile 128x64,
// wave tile 32x32 (2x2 WMMA register blocking)
// ---------------------------------------------------------------------------
template<int ACT>  // 0 none, 1 sigmoid, 2 relu
__global__ __launch_bounds__(256)
void gemm_bf16o_k(const __bf16* __restrict__ A, int ldA,
                  const __bf16* __restrict__ Bt, int ldB,
                  const float* __restrict__ bias,
                  __bf16* __restrict__ out, int N, int K) {
  int w = threadIdx.x >> 5, lane = threadIdx.x & 31, half = lane >> 4, l16 = lane & 15;
  int rowBase = blockIdx.y * 128 + (w & 3) * 32;
  int colBase = blockIdx.x * 64 + (w >> 2) * 32;
  Acc4 a4 = mma_loop2(A, ldA, rowBase, Bt, ldB, colBase, K);
  for (int t = 0; t < 4; ++t) {
    floatx8 acc = acc_sel(a4, t);
    int tm = t >> 1, tn = t & 1;
    for (int r = 0; r < 8; ++r) {
      int grow = rowBase + tm * 16 + half * 8 + r;
      int gcol = colBase + tn * 16 + l16;
      float v = acc[r] + (bias ? bias[gcol] : 0.f);
      if (ACT == 1) v = sigmoidf_(v);
      if (ACT == 2) v = fmaxf(v, 0.f);
      out[(size_t)grow * N + gcol] = f2bf(v);
    }
  }
}

__global__ __launch_bounds__(256)
void gemm_f32o_k(const __bf16* __restrict__ A, int ldA,
                 const __bf16* __restrict__ Bt, int ldB,
                 const float* __restrict__ bias,
                 float* __restrict__ out, int N, int K) {
  int w = threadIdx.x >> 5, lane = threadIdx.x & 31, half = lane >> 4, l16 = lane & 15;
  int rowBase = blockIdx.y * 128 + (w & 3) * 32;
  int colBase = blockIdx.x * 64 + (w >> 2) * 32;
  Acc4 a4 = mma_loop2(A, ldA, rowBase, Bt, ldB, colBase, K);
  for (int t = 0; t < 4; ++t) {
    floatx8 acc = acc_sel(a4, t);
    int tm = t >> 1, tn = t & 1;
    for (int r = 0; r < 8; ++r) {
      int grow = rowBase + tm * 16 + half * 8 + r;
      int gcol = colBase + tn * 16 + l16;
      out[(size_t)grow * N + gcol] = acc[r] + bias[gcol];
    }
  }
}

// a/b = mask * sigmoid(zl@Wg + bg) * pre  -> bf16 channel-major [c][i][k]
// (transIK swaps i/k for the "incoming" triangle variant)
__global__ __launch_bounds__(256)
void gemm_gatemul_k(const __bf16* __restrict__ A, int ldA,
                    const __bf16* __restrict__ Bt, int ldB,
                    const float* __restrict__ bias,
                    const float* __restrict__ pre,
                    const float* __restrict__ mask,
                    __bf16* __restrict__ outCh, int K, int transIK) {
  int w = threadIdx.x >> 5, lane = threadIdx.x & 31, half = lane >> 4, l16 = lane & 15;
  int rowBase = blockIdx.y * 128 + (w & 3) * 32;
  int colBase = blockIdx.x * 64 + (w >> 2) * 32;
  Acc4 a4 = mma_loop2(A, ldA, rowBase, Bt, ldB, colBase, K);
  for (int t = 0; t < 4; ++t) {
    floatx8 acc = acc_sel(a4, t);
    int tm = t >> 1, tn = t & 1;
    for (int r = 0; r < 8; ++r) {
      int grow = rowBase + tm * 16 + half * 8 + r;
      int gcol = colBase + tn * 16 + l16;
      float v = sigmoidf_(acc[r] + bias[gcol]) * pre[(size_t)grow * CH + gcol] * mask[grow];
      int i = grow >> 8, kk = grow & 255;
      size_t dst = (size_t)gcol * MROWS +
                   (transIK ? ((size_t)kk * 256 + i) : ((size_t)i * 256 + kk));
      outCh[dst] = f2bf(v);
    }
  }
}

// batched triangle einsum: X[i,j,c] = sum_t Ach[c][i][t] * Bch[c][j][t]
__global__ __launch_bounds__(256)
void tri_einsum_k(const __bf16* __restrict__ Ach, const __bf16* __restrict__ Bch,
                  float* __restrict__ X) {
  int c = blockIdx.z;
  const __bf16* A  = Ach + (size_t)c * MROWS;
  const __bf16* Bt = Bch + (size_t)c * MROWS;
  int w = threadIdx.x >> 5, lane = threadIdx.x & 31, half = lane >> 4, l16 = lane & 15;
  int rowBase = blockIdx.y * 128 + (w & 3) * 32;
  int colBase = blockIdx.x * 64 + (w >> 2) * 32;
  Acc4 a4 = mma_loop2(A, NRES, rowBase, Bt, NRES, colBase, NRES);
  for (int t = 0; t < 4; ++t) {
    floatx8 acc = acc_sel(a4, t);
    int tm = t >> 1, tn = t & 1;
    for (int r = 0; r < 8; ++r) {
      int i = rowBase + tm * 16 + half * 8 + r;
      int j = colBase + tn * 16 + l16;
      X[(size_t)(i * 256 + j) * CH + c] = acc[r];
    }
  }
}

// final residual GEMMs into z (fp32, N=384). MODE 0: gated; 1: plain; 2: masked.
template<int MODE>
__global__ __launch_bounds__(256)
void gemm_resid_k(const __bf16* __restrict__ A, int ldA,
                  const __bf16* __restrict__ Bt, int ldB,
                  const float* __restrict__ bias,
                  const __bf16* __restrict__ gate,
                  const float* __restrict__ rmask,
                  float* __restrict__ zio, int K, int transT) {
  int w = threadIdx.x >> 5, lane = threadIdx.x & 31, half = lane >> 4, l16 = lane & 15;
  int rowBase = blockIdx.y * 128 + (w & 3) * 32;
  int colBase = blockIdx.x * 64 + (w >> 2) * 32;
  Acc4 a4 = mma_loop2(A, ldA, rowBase, Bt, ldB, colBase, K);
  for (int t = 0; t < 4; ++t) {
    floatx8 acc = acc_sel(a4, t);
    int tm = t >> 1, tn = t & 1;
    for (int r = 0; r < 8; ++r) {
      int grow = rowBase + tm * 16 + half * 8 + r;
      int gcol = colBase + tn * 16 + l16;
      float v = acc[r] + bias[gcol];
      if (MODE == 0) v *= bf2f(gate[(size_t)grow * CS + gcol]);
      if (MODE == 2) v *= rmask[grow];
      int trow = transT ? (((grow & 255) << 8) | (grow >> 8)) : grow;
      zio[(size_t)trow * CS + gcol] += v;
    }
  }
}

// ---------------------------------------------------------------------------
// Fused triangle attention: one block per (h, i, 32-row j tile)
// ---------------------------------------------------------------------------
__global__ __launch_bounds__(256)
void tri_attn_k(const __bf16* __restrict__ Q, const __bf16* __restrict__ Kb,
                const __bf16* __restrict__ Vb, const float* __restrict__ trib,
                const __bf16* __restrict__ gate, const float* __restrict__ mask,
                __bf16* __restrict__ AO) {
  __shared__ __bf16 Vt[32 * 256];   // V transposed [d][k]
  __shared__ float  S[32 * 256];    // scores (later overlaid by bf16 P)
  __shared__ float  mb[256];
  __shared__ float  red[32 * 8];
  int jt = blockIdx.x, i = blockIdx.y, h = blockIdx.z;
  int tid = threadIdx.x;
  {
    int kk = tid;                                   // 0..255
    const __bf16* vr = Vb + ((size_t)(i * 256 + kk) * CH + h * 32);
    for (int d = 0; d < 32; ++d) Vt[d * 256 + kk] = vr[d];
    mb[kk] = (mask[i * 256 + kk] - 1.f) * 1e9f;
  }
  __syncthreads();

  int w = tid >> 5, lane = tid & 31, half = lane >> 4, l16 = lane & 15;
  int jm = w & 1, kn = w >> 1;                      // 2 x 4 wave grid over S
  const float scale = 0.17677669529663687f;        // 1/sqrt(32)
  for (int t = 0; t < 4; ++t) {
    int col0 = kn * 64 + t * 16;
    const __bf16* pa = Q  + ((size_t)(i * 256 + jt * 32 + jm * 16 + l16) * CH + h * 32 + half * 8);
    const __bf16* pb = Kb + ((size_t)(i * 256 + col0 + l16) * CH + h * 32 + half * 8);
    floatx8 z8 = {0.f,0.f,0.f,0.f,0.f,0.f,0.f,0.f};
    floatx8 acc = wmma_bf16(load_fragA(pa), load_fragB(pb), z8);
    for (int r = 0; r < 8; ++r) {
      int jl = jm * 16 + half * 8 + r;
      int kcol = col0 + l16;
      int j = jt * 32 + jl;
      S[jl * 256 + kcol] = acc[r] * scale + trib[((size_t)j * 256 + kcol) * 4 + h] + mb[kcol];
    }
  }
  __syncthreads();

  // softmax: 32 rows x 8 segments of 32 cols each
  int rrow = tid >> 3, seg = tid & 7;
  float* srow = S + rrow * 256 + seg * 32;
  float mx = -3.0e38f;
  for (int c = 0; c < 32; ++c) mx = fmaxf(mx, srow[c]);
  red[rrow * 8 + seg] = mx;
  __syncthreads();
  float rmax = red[rrow * 8];
  for (int c = 1; c < 8; ++c) rmax = fmaxf(rmax, red[rrow * 8 + c]);
  float ev[32]; float sum = 0.f;
  for (int c = 0; c < 32; ++c) { float e = __expf(srow[c] - rmax); ev[c] = e; sum += e; }
  __syncthreads();
  red[rrow * 8 + seg] = sum;
  __syncthreads();
  float rsum = 0.f;
  for (int c = 0; c < 8; ++c) rsum += red[rrow * 8 + c];
  float inv = 1.f / rsum;
  __bf16* P = (__bf16*)S;                          // overlay S (reads done)
  for (int c = 0; c < 32; ++c) P[rrow * 256 + seg * 32 + c] = f2bf(ev[c] * inv);
  __syncthreads();

  // O = P(32x256) @ Vt(32x256)^T : waves 0..3, 2x2 wave grid
  if (w < 4) {
    int jm2 = w & 1, dn = w >> 1;
    floatx8 acc = {0.f,0.f,0.f,0.f,0.f,0.f,0.f,0.f};
    const __bf16* pa = P  + ((size_t)(jm2 * 16 + l16) * 256 + half * 8);
    const __bf16* pb = Vt + ((size_t)(dn * 16 + l16) * 256 + half * 8);
    for (int k0 = 0; k0 < 256; k0 += 32)
      acc = wmma_bf16(load_fragA(pa + k0), load_fragB(pb + k0), acc);
    for (int r = 0; r < 8; ++r) {
      int j = jt * 32 + jm2 * 16 + half * 8 + r;
      int d = dn * 16 + l16;
      size_t o = (size_t)(i * 256 + j) * CH + h * 32 + d;
      AO[o] = f2bf(acc[r] * bf2f(gate[o]));
    }
  }
}

// ---------------------------------------------------------------------------
// Host orchestration
// ---------------------------------------------------------------------------
extern "C" void kernel_launch(void* const* d_in, const int* in_sizes, int n_in,
                              void* d_out, int out_size, void* d_ws, size_t ws_size,
                              hipStream_t stream) {
  (void)in_sizes; (void)n_in; (void)out_size; (void)ws_size;
  const float* zin  = (const float*)d_in[0];
  const float* mask = (const float*)d_in[1];
  float* zcur = (float*)d_out;                 // running z (fp32, 100 MB)
  char*  ws   = (char*)d_ws;

  // workspace layout (requires ~257 MB):
  //   [0, 50 MB)   : LNZ bf16 [M,384]
  //   [50, 251 MB) : phase-shared region P0
  //   [251 MB ...) : bf16-transposed weights
  __bf16* LNZ = (__bf16*)ws;
  char* P0 = ws + (size_t)50331648;
  size_t wofs = (size_t)50331648 + (size_t)201326592;

  auto cvt = [&](int idx, int In, int Out) -> const __bf16* {
    __bf16* p = (__bf16*)(ws + wofs);
    size_t n = (size_t)In * Out;
    wofs += ((n * 2 + 255) / 256) * 256;
    int total = (int)n;
    wt_cast_k<<<dim3((total + 255) / 256), dim3(256), 0, stream>>>(
        (const float*)d_in[idx], p, In, Out);
    return p;
  };

  (void)hipMemcpyAsync(zcur, zin, (size_t)MROWS * CS * sizeof(float),
                       hipMemcpyDeviceToDevice, stream);

  // weights (input dict order: z, mask, tmo(16), tmi(16), tas(10), tae(10), pt(6))
  const __bf16 *tmo_ap = cvt(4, 384, 128),  *tmo_ag = cvt(6, 384, 128);
  const __bf16 *tmo_bp = cvt(8, 384, 128),  *tmo_bg = cvt(10, 384, 128);
  const __bf16 *tmo_g  = cvt(14, 384, 384), *tmo_z  = cvt(16, 128, 384);
  const __bf16 *tmi_ap = cvt(20, 384, 128), *tmi_ag = cvt(22, 384, 128);
  const __bf16 *tmi_bp = cvt(24, 384, 128), *tmi_bg = cvt(26, 384, 128);
  const __bf16 *tmi_g  = cvt(30, 384, 384), *tmi_z  = cvt(32, 128, 384);
  const __bf16 *tas_q = cvt(36, 384, 128), *tas_k = cvt(37, 384, 128);
  const __bf16 *tas_v = cvt(38, 384, 128), *tas_g = cvt(40, 384, 128);
  const __bf16 *tas_o = cvt(42, 128, 384);
  const __bf16 *tae_q = cvt(46, 384, 128), *tae_k = cvt(47, 384, 128);
  const __bf16 *tae_v = cvt(48, 384, 128), *tae_g = cvt(50, 384, 128);
  const __bf16 *tae_o = cvt(52, 128, 384);
  const __bf16 *pt_w1 = cvt(56, 384, 1536), *pt_w2 = cvt(58, 1536, 384);

  // ---- triangle multiplication ----
  float*  APf  = (float*)(P0);                     // 33.5 MB scratch (pre-act)
  __bf16* ACH  = (__bf16*)(P0 + 33554432);         // a channel-major
  __bf16* BCH  = (__bf16*)(P0 + 50331648);         // b channel-major
  float*  Xf   = (float*)(P0 + 67108864);          // einsum result fp32
  __bf16* GATE = (__bf16*)(P0 + 100663296);        // gate bf16 [M,384]
  __bf16* XL   = (__bf16*)(P0);                    // LN(x) bf16 (reuses APf)

  auto tri_mult = [&](int base, const __bf16* Wap, const __bf16* Wag,
                      const __bf16* Wbp, const __bf16* Wbg,
                      const __bf16* Wg, const __bf16* Wz, bool outgoing) {
    const float* ln_in_g  = (const float*)d_in[base + 0];
    const float* ln_in_b  = (const float*)d_in[base + 1];
    const float* apb      = (const float*)d_in[base + 3];
    const float* agb      = (const float*)d_in[base + 5];
    const float* bpb      = (const float*)d_in[base + 7];
    const float* bgb      = (const float*)d_in[base + 9];
    const float* ln_out_g = (const float*)d_in[base + 10];
    const float* ln_out_b = (const float*)d_in[base + 11];
    const float* gb       = (const float*)d_in[base + 13];
    const float* zb       = (const float*)d_in[base + 15];
    int trans = outgoing ? 0 : 1;
    ln_bf16_k<<<MROWS, 128, 0, stream>>>(zcur, ln_in_g, ln_in_b, LNZ, CS);
    gemm_f32o_k<<<dim3(2, 512), 256, 0, stream>>>(LNZ, CS, Wap, CS, apb, APf, CH, CS);
    gemm_gatemul_k<<<dim3(2, 512), 256, 0, stream>>>(LNZ, CS, Wag, CS, agb, APf, mask,
                                                     ACH, CS, trans);
    gemm_f32o_k<<<dim3(2, 512), 256, 0, stream>>>(LNZ, CS, Wbp, CS, bpb, APf, CH, CS);
    gemm_gatemul_k<<<dim3(2, 512), 256, 0, stream>>>(LNZ, CS, Wbg, CS, bgb, APf, mask,
                                                     BCH, CS, trans);
    tri_einsum_k<<<dim3(4, 2, 128), 256, 0, stream>>>(ACH, BCH, Xf);
    ln_bf16_k<<<MROWS, 128, 0, stream>>>(Xf, ln_out_g, ln_out_b, XL, CH);
    gemm_bf16o_k<1><<<dim3(6, 512), 256, 0, stream>>>(LNZ, CS, Wg, CS, gb, GATE, CS, CS);
    gemm_resid_k<0><<<dim3(6, 512), 256, 0, stream>>>(XL, CH, Wz, CH, zb, GATE, nullptr,
                                                      zcur, CH, 0);
  };

  // ---- triangle attention ----
  __bf16* Qb    = (__bf16*)(P0);
  __bf16* Kbuf  = (__bf16*)(P0 + 16777216);
  __bf16* Vbuf  = (__bf16*)(P0 + 33554432);
  float*  TRIB  = (float*)(P0 + 50331648);
  __bf16* GATEA = (__bf16*)(P0 + 52428800);
  __bf16* AO    = (__bf16*)(P0 + 69206016);
  float*  ZT    = (float*)(P0 + 85983232);
  float*  MASKT = (float*)(P0 + 186646528);

  auto tri_attn = [&](int base, const __bf16* Wq, const __bf16* Wk,
                      const __bf16* Wv, const __bf16* Wg, const __bf16* Wo,
                      bool transposed) {
    const float* lng = (const float*)d_in[base + 0];
    const float* lnb = (const float*)d_in[base + 1];
    const float* bw  = (const float*)d_in[base + 5];
    const float* gb  = (const float*)d_in[base + 7];
    const float* ob  = (const float*)d_in[base + 9];
    const float* zsrc = zcur; const float* msrc = mask;
    if (transposed) {
      transpose_z_k<<<dim3(98304), 256, 0, stream>>>(zcur, ZT);
      mask_t_k<<<dim3(256), 256, 0, stream>>>(mask, MASKT);
      zsrc = ZT; msrc = MASKT;
    }
    ln_bf16_k<<<MROWS, 128, 0, stream>>>(zsrc, lng, lnb, LNZ, CS);
    gemm_bf16o_k<0><<<dim3(2, 512), 256, 0, stream>>>(LNZ, CS, Wq, CS, nullptr, Qb, CH, CS);
    gemm_bf16o_k<0><<<dim3(2, 512), 256, 0, stream>>>(LNZ, CS, Wk, CS, nullptr, Kbuf, CH, CS);
    gemm_bf16o_k<0><<<dim3(2, 512), 256, 0, stream>>>(LNZ, CS, Wv, CS, nullptr, Vbuf, CH, CS);
    proj4_k<<<dim3(1024), 256, 0, stream>>>(LNZ, bw, TRIB);
    gemm_bf16o_k<1><<<dim3(2, 512), 256, 0, stream>>>(LNZ, CS, Wg, CS, gb, GATEA, CH, CS);
    tri_attn_k<<<dim3(8, 256, 4), 256, 0, stream>>>(Qb, Kbuf, Vbuf, TRIB, GATEA, msrc, AO);
    gemm_resid_k<1><<<dim3(6, 512), 256, 0, stream>>>(AO, CH, Wo, CH, ob, nullptr, nullptr,
                                                      zcur, CH, transposed ? 1 : 0);
  };

  // ---- pipeline ----
  tri_mult(2,  tmo_ap, tmo_ag, tmo_bp, tmo_bg, tmo_g, tmo_z, true);
  tri_mult(18, tmi_ap, tmi_ag, tmi_bp, tmi_bg, tmi_g, tmi_z, false);
  tri_attn(34, tas_q, tas_k, tas_v, tas_g, tas_o, false);
  tri_attn(44, tae_q, tae_k, tae_v, tae_g, tae_o, true);

  // ---- pair transition ----
  __bf16* HID = (__bf16*)(P0);                 // [M,1536] bf16
  ln_bf16_k<<<MROWS, 128, 0, stream>>>(zcur, (const float*)d_in[54],
                                       (const float*)d_in[55], LNZ, CS);
  gemm_bf16o_k<2><<<dim3(24, 512), 256, 0, stream>>>(LNZ, CS, pt_w1, CS,
                                                     (const float*)d_in[57], HID, 1536, CS);
  gemm_resid_k<2><<<dim3(6, 512), 256, 0, stream>>>(HID, 1536, pt_w2, 1536,
                                                    (const float*)d_in[59], nullptr, mask,
                                                    zcur, 1536, 0);
}
